// GCNNetSortPooling_41120016892599
// MI455X (gfx1250) — matrix operand order repaired and checked by reading, not compiled, not run on hardware
//
#include <hip/hip_runtime.h>
#include <hip/hip_bf16.h>

#define NN    100000
#define NE    3200000
#define NPER  400
#define NG    250
#define KTOP  100
#define GDIM  128
#define DENSE 2944

typedef __attribute__((ext_vector_type(16))) _Float16 v16h;
typedef __attribute__((ext_vector_type(8)))  float    v8f;

// ---- 16-bit A-matrix 16x32 lane layout (ISA 7.12.2): VGPR v pair-base K ----
__device__ __forceinline__ int kbaseA(int v) { return (v < 4) ? (v * 2) : (16 + (v - 4) * 2); }

union AFrag { v16h v; _Float16 h[16]; unsigned int u[8]; };

// ---------------------------------------------------------------- utilities
__global__ void k_fill_f32(float* __restrict__ p, long long n, float v) {
  long long i = (long long)blockIdx.x * blockDim.x + threadIdx.x;
  if (i < n) p[i] = v;
}

__global__ void k_f32_to_f16(const float* __restrict__ in, _Float16* __restrict__ out, long long n) {
  long long i = (long long)blockIdx.x * blockDim.x + threadIdx.x;
  if (i < n) out[i] = (_Float16)in[i];
}

// ---------------------------------------------------------------- degrees
__global__ void k_deg_count(const int* __restrict__ dst, float* __restrict__ deg, int E) {
  int e = blockIdx.x * blockDim.x + threadIdx.x;
  if (e < E) atomicAdd(&deg[dst[e]], 1.0f);
}

__global__ void k_norms(const float* __restrict__ deg, float* __restrict__ dis,
                        float* __restrict__ deginv, int n) {
  int i = blockIdx.x * blockDim.x + threadIdx.x;
  if (i < n) { float d = deg[i]; dis[i] = rsqrtf(d); deginv[i] = 1.0f / d; }
}

// ------------------------------------------------- B-fragment prep (weights)
// Packs B[K,N] (various source layouts) into WMMA-B fragment order:
// frag[((ks*ntiles+nt)*32 + lane)*16 + j], with k = ks*32 + (j<8 ? (lane>>4)*8 + j : 16 + (lane>>4)*8 + j-8)
// (B 32x16 16-bit: lanes 0-15 hold K=0..7,16..23, lanes 16-31 K=8..15,24..31, 2/VGPR).
__global__ void k_build_frag(const float* __restrict__ W, _Float16* __restrict__ frag,
                             int Kreal, int ksteps, int ntiles, int Ncols, int mode) {
  int t = blockIdx.x * blockDim.x + threadIdx.x;
  int total = ksteps * ntiles * 512;
  if (t >= total) return;
  int j = t & 15, lane = (t >> 4) & 31, rest = t >> 9;
  int nt = rest % ntiles, ks = rest / ntiles;
  int k = ks * 32 + ((j < 8) ? ((lane >> 4) * 8 + j) : (16 + (lane >> 4) * 8 + (j - 8)));
  int n = nt * 16 + (lane & 15);
  float val = 0.0f;
  if (k < Kreal && n < Ncols) {
    size_t idx;
    if (mode == 0)      idx = (size_t)k * Ncols + n;          // W[K,N] row-major
    else if (mode == 1) idx = (size_t)n * 97 + k;             // cw1[c,0,t] -> B[t][c]
    else if (mode == 2) { int r = k >> 7, i = k & 127;        // cw2[o,i,r] -> B[r*128+i][o]
                          idx = (size_t)n * 640 + i * 5 + r; }
    else                idx = (size_t)n * 2944 + k;           // fw1[g,j] -> B[j][g]
    val = W[idx];
  }
  frag[t] = (_Float16)val;
}

// ---------------------------------------------------------------- WMMA GEMM
// One wave computes a 16x16 tile of C = A(f16)[M,K] @ Bfrag(f16)[K,N].
// mode 0: raw store C[row*ldc+n]
// mode 1: relu(x + bias[n]) store C[row*ldc+n]
// mode 2: conv2: A row m -> (b=m/46,h=m%46), arow = A + b*6400 + h*128 (640 contiguous);
//         store relu(x+bias[n]) at C[b*2944 + n*46 + h]
__global__ void k_gemm_wmma(const _Float16* __restrict__ A, const _Float16* __restrict__ Bfrag,
                            float* __restrict__ C, const float* __restrict__ bias,
                            int Mreal, int lda, int ksteps, int ntiles, int mtiles,
                            int ldc, int mode) {
  int wave = blockIdx.x * (blockDim.x >> 5) + (threadIdx.x >> 5);
  int lane = threadIdx.x & 31;
  if (wave >= mtiles * ntiles) return;
  int mtile = wave / ntiles, ntile = wave % ntiles;
  int khalf = lane >> 4;
  int arowi = mtile * 16 + (lane & 15);
  bool arowOK = (arowi < Mreal);
  const _Float16* arow = nullptr;
  if (arowOK) {
    if (mode == 2) { int b = arowi / 46, hh = arowi % 46; arow = A + (size_t)b * 6400 + (size_t)hh * 128; }
    else           { arow = A + (size_t)arowi * lda; }
  }
  const _Float16* bp = Bfrag + ((size_t)ntile * 32 + lane) * 16;
  v8f acc = {0.f, 0.f, 0.f, 0.f, 0.f, 0.f, 0.f, 0.f};
  for (int ks = 0; ks < ksteps; ++ks) {
    AFrag a;
    if (arowOK) {
      int k0 = ks * 32 + khalf * 8;
#pragma unroll
      for (int v = 0; v < 8; ++v) a.u[v] = *(const unsigned int*)(arow + k0 + kbaseA(v));
    } else {
#pragma unroll
      for (int v = 0; v < 8; ++v) a.u[v] = 0u;
    }
    AFrag b; b.v = *(const v16h*)bp;
    bp += (size_t)ntiles * 512;
    acc = __builtin_amdgcn_wmma_f32_16x16x32_f16(false, a.v, false, b.v, (short)0, acc, false, false);
  }
  int ncol = ntile * 16 + (lane & 15);
#pragma unroll
  for (int r = 0; r < 8; ++r) {            // D: VGPR r -> row r + 8*khalf
    int srow = mtile * 16 + r + khalf * 8;
    if (srow >= Mreal) continue;
    float val = acc[r];
    if (mode == 0) {
      C[(size_t)srow * ldc + ncol] = val;
    } else if (mode == 1) {
      C[(size_t)srow * ldc + ncol] = fmaxf(val + bias[ncol], 0.0f);
    } else {
      int b = srow / 46, hh = srow % 46;
      C[(size_t)b * 2944 + (size_t)ncol * 46 + hh] = fmaxf(val + bias[ncol], 0.0f);
    }
  }
}

// ---------------------------------------------------------------- GCN pieces
__global__ void k_scatter32(const int* __restrict__ src, const int* __restrict__ dst,
                            const float* __restrict__ dis, const float* __restrict__ hlin,
                            float* __restrict__ agg, int E) {
  long long t = (long long)blockIdx.x * blockDim.x + threadIdx.x;
  int e = (int)(t >> 3);
  if (e >= E) return;
  int q = (int)(t & 7);
  int s = src[e], d = dst[e];
  float c = dis[s] * dis[d];
  float4 v = *(const float4*)(hlin + (size_t)s * 32 + q * 4);
  float* ap = agg + (size_t)d * 32 + q * 4;
  atomicAdd(ap + 0, v.x * c); atomicAdd(ap + 1, v.y * c);
  atomicAdd(ap + 2, v.z * c); atomicAdd(ap + 3, v.w * c);
}

__global__ void k_scatter1(const int* __restrict__ src, const int* __restrict__ dst,
                           const float* __restrict__ dis, const float* __restrict__ hlin1,
                           float* __restrict__ agg1, int E) {
  int e = blockIdx.x * blockDim.x + threadIdx.x;
  if (e >= E) return;
  int s = src[e], d = dst[e];
  atomicAdd(&agg1[d], hlin1[s] * dis[s] * dis[d]);
}

__global__ void k_lin32to1(const float* __restrict__ hin, const float* __restrict__ W4,
                           float* __restrict__ out, int n) {
  int i = blockIdx.x * blockDim.x + threadIdx.x;
  if (i >= n) return;
  float s = 0.f;
#pragma unroll
  for (int k = 0; k < 32; ++k) s += hin[(size_t)i * 32 + k] * W4[k];
  out[i] = s;
}

__global__ void k_combine(const float* __restrict__ agg, const float* __restrict__ hlin,
                          const float* __restrict__ deginv, const float* __restrict__ bias,
                          float* __restrict__ out, long long n, int C) {
  long long t = (long long)blockIdx.x * blockDim.x + threadIdx.x;
  if (t >= n) return;
  long long i = t / C; int c = (int)(t % C);
  out[t] = tanhf(agg[t] + hlin[t] * deginv[i] + bias[c]);
}

// ---------------------------------------------------------------- sort-pool
__device__ __forceinline__ unsigned int flipf(float f) {
  unsigned int u = __float_as_uint(f);
  return (u & 0x80000000u) ? ~u : (u | 0x80000000u);
}

// One block per graph: stable descending sort of 400 keys (bitonic over 512),
// gather top-100 rows of [x1|x2|x3|x4] directly into f16 GEMM-A (K padded to 128).
__global__ void k_sortpool(const float* __restrict__ h1, const float* __restrict__ h2,
                           const float* __restrict__ h3, const float* __restrict__ h4,
                           _Float16* __restrict__ xs16) {
  __shared__ unsigned long long s[512];
  __shared__ int nodes[KTOP];
  int g = blockIdx.x, tid = threadIdx.x;
  float key = -3.0e38f;
  if (tid < NPER) key = h4[g * NPER + tid];
  s[tid] = ((unsigned long long)(~flipf(key)) << 32) | (unsigned int)tid;
  __syncthreads();
  for (int k = 2; k <= 512; k <<= 1) {
    for (int j = k >> 1; j > 0; j >>= 1) {
      int ixj = tid ^ j;
      if (ixj > tid) {
        unsigned long long a = s[tid], b = s[ixj];
        bool up = (tid & k) == 0;
        if ((a > b) == up) { s[tid] = b; s[ixj] = a; }
      }
      __syncthreads();
    }
  }
  if (tid < KTOP) nodes[tid] = g * NPER + (int)(s[tid] & 0x1FFu);
  __syncthreads();
  for (int e = tid; e < KTOP * 128; e += blockDim.x) {
    int k = e >> 7, c = e & 127;
    int node = nodes[k];
    float v = 0.f;
    if (c < 32)       v = h1[(size_t)node * 32 + c];
    else if (c < 64)  v = h2[(size_t)node * 32 + (c - 32)];
    else if (c < 96)  v = h3[(size_t)node * 32 + (c - 64)];
    else if (c == 96) v = h4[node];
    xs16[((size_t)g * KTOP + k) * 128 + c] = (_Float16)v;
  }
}

// MaxPool1d(2,2) over k, output f16 [b][j][c] for conv2 GEMM-A windows
__global__ void k_pool(const float* __restrict__ c1, _Float16* __restrict__ p1, long long n) {
  long long t = (long long)blockIdx.x * blockDim.x + threadIdx.x;
  if (t >= n) return;
  int c = (int)(t & 127);
  int j = (int)((t >> 7) % 50);
  int b = (int)(t / (128 * 50));
  float a0 = c1[((size_t)b * 100 + 2 * j) * 128 + c];
  float a1 = c1[((size_t)b * 100 + 2 * j + 1) * 128 + c];
  p1[t] = (_Float16)fmaxf(a0, a1);
}

// fc3 + log_softmax
__global__ void k_fc3(const float* __restrict__ h, const float* __restrict__ fw3,
                      const float* __restrict__ fb3, float* __restrict__ out) {
  int b = blockIdx.x * blockDim.x + threadIdx.x;
  if (b >= NG) return;
  float z0 = fb3[0], z1 = fb3[1];
  for (int k = 0; k < GDIM; ++k) {
    float hv = h[(size_t)b * GDIM + k];
    z0 += hv * fw3[k];
    z1 += hv * fw3[GDIM + k];
  }
  float m = fmaxf(z0, z1);
  float lse = m + logf(expf(z0 - m) + expf(z1 - m));
  out[b * 2 + 0] = z0 - lse;
  out[b * 2 + 1] = z1 - lse;
}

// ================================================================= launcher
static inline int divup(long long a, int b) { return (int)((a + b - 1) / b); }

extern "C" void kernel_launch(void* const* d_in, const int* in_sizes, int n_in,
                              void* d_out, int out_size, void* d_ws, size_t ws_size,
                              hipStream_t stream) {
  const float* x   = (const float*)d_in[0];
  const int*   ei  = (const int*)d_in[1];
  const int*   src = ei;
  const int*   dst = ei + NE;
  const float* W1 = (const float*)d_in[3];  const float* b1 = (const float*)d_in[4];
  const float* W2 = (const float*)d_in[5];  const float* b2 = (const float*)d_in[6];
  const float* W3 = (const float*)d_in[7];  const float* b3 = (const float*)d_in[8];
  const float* W4 = (const float*)d_in[9];  const float* b4 = (const float*)d_in[10];
  const float* cw1 = (const float*)d_in[11]; const float* cb1 = (const float*)d_in[12];
  const float* cw2 = (const float*)d_in[13]; const float* cb2 = (const float*)d_in[14];
  const float* fw1 = (const float*)d_in[15]; const float* fb1 = (const float*)d_in[16];
  const float* fw3 = (const float*)d_in[17]; const float* fb3 = (const float*)d_in[18];
  float* out = (float*)d_out;

  char* ws = (char*)d_ws;
  size_t off = 0;
  auto alloc = [&](size_t bytes) -> void* {
    void* p = ws + off;
    off = (off + bytes + 255) & ~(size_t)255;
    return p;
  };

  float* deg    = (float*)alloc((size_t)NN * 4);
  float* dis    = (float*)alloc((size_t)NN * 4);
  float* deginv = (float*)alloc((size_t)NN * 4);
  float* hlin   = (float*)alloc((size_t)NN * 32 * 4);
  float* agg    = (float*)alloc((size_t)NN * 32 * 4);
  float* h1     = (float*)alloc((size_t)NN * 32 * 4);
  float* h2     = (float*)alloc((size_t)NN * 32 * 4);
  float* h3     = (float*)alloc((size_t)NN * 32 * 4);
  float* hlin1  = (float*)alloc((size_t)NN * 4);
  float* agg1   = (float*)alloc((size_t)NN * 4);
  float* h4     = (float*)alloc((size_t)NN * 4);
  _Float16* xf16   = (_Float16*)alloc((size_t)NN * 128 * 2);
  _Float16* hf16   = (_Float16*)alloc((size_t)NN * 32 * 2);
  _Float16* h2f16  = (_Float16*)alloc((size_t)NN * 32 * 2);
  _Float16* xs16   = (_Float16*)alloc((size_t)NG * KTOP * 128 * 2);
  float*    c1     = (float*)alloc((size_t)NG * KTOP * 128 * 4);
  _Float16* p1f16  = (_Float16*)alloc((size_t)NG * 50 * 128 * 2);
  float*    c2     = (float*)alloc((size_t)NG * DENSE * 4);
  _Float16* c2f16  = (_Float16*)alloc((size_t)NG * DENSE * 2);
  float*    fco    = (float*)alloc((size_t)NG * GDIM * 4);
  _Float16* w1f  = (_Float16*)alloc((size_t)4  * 2 * 512 * 2);
  _Float16* w2f  = (_Float16*)alloc((size_t)1  * 2 * 512 * 2);
  _Float16* w3f  = (_Float16*)alloc((size_t)1  * 2 * 512 * 2);
  _Float16* cw1f = (_Float16*)alloc((size_t)4  * 8 * 512 * 2);
  _Float16* cw2f = (_Float16*)alloc((size_t)20 * 4 * 512 * 2);
  _Float16* fw1f = (_Float16*)alloc((size_t)92 * 8 * 512 * 2);
  (void)ws_size; (void)in_sizes; (void)n_in; (void)out_size;

  const int TB = 256;

  // ---- weight fragment prep
  k_build_frag<<<divup(4 * 2 * 512, TB), TB, 0, stream>>>(W1,  w1f,  128,  4, 2, 32,  0);
  k_build_frag<<<divup(1 * 2 * 512, TB), TB, 0, stream>>>(W2,  w2f,  32,   1, 2, 32,  0);
  k_build_frag<<<divup(1 * 2 * 512, TB), TB, 0, stream>>>(W3,  w3f,  32,   1, 2, 32,  0);
  k_build_frag<<<divup(4 * 8 * 512, TB), TB, 0, stream>>>(cw1, cw1f, 97,   4, 8, 128, 1);
  k_build_frag<<<divup(20 * 4 * 512, TB), TB, 0, stream>>>(cw2, cw2f, 640, 20, 4, 64, 2);
  k_build_frag<<<divup(92 * 8 * 512, TB), TB, 0, stream>>>(fw1, fw1f, 2944, 92, 8, 128, 3);

  // ---- degrees / norms
  k_fill_f32<<<divup(NN, TB), TB, 0, stream>>>(deg, NN, 1.0f);
  k_deg_count<<<divup(NE, TB), TB, 0, stream>>>(dst, deg, NE);
  k_norms<<<divup(NN, TB), TB, 0, stream>>>(deg, dis, deginv, NN);

  // ---- A conversion for layer 1
  k_f32_to_f16<<<divup((long long)NN * 128, TB), TB, 0, stream>>>(x, xf16, (long long)NN * 128);

  // ---- GCN layer 1: x @ W1
  k_gemm_wmma<<<divup(6250 * 2, 8), TB, 0, stream>>>(xf16, w1f, hlin, nullptr, NN, 128, 4, 2, 6250, 32, 0);
  k_fill_f32<<<divup((long long)NN * 32, TB), TB, 0, stream>>>(agg, (long long)NN * 32, 0.0f);
  k_scatter32<<<divup((long long)NE * 8, TB), TB, 0, stream>>>(src, dst, dis, hlin, agg, NE);
  k_combine<<<divup((long long)NN * 32, TB), TB, 0, stream>>>(agg, hlin, deginv, b1, h1, (long long)NN * 32, 32);
  k_f32_to_f16<<<divup((long long)NN * 32, TB), TB, 0, stream>>>(h1, hf16, (long long)NN * 32);

  // ---- GCN layer 2: x1 @ W2
  k_gemm_wmma<<<divup(6250 * 2, 8), TB, 0, stream>>>(hf16, w2f, hlin, nullptr, NN, 32, 1, 2, 6250, 32, 0);
  k_fill_f32<<<divup((long long)NN * 32, TB), TB, 0, stream>>>(agg, (long long)NN * 32, 0.0f);
  k_scatter32<<<divup((long long)NE * 8, TB), TB, 0, stream>>>(src, dst, dis, hlin, agg, NE);
  k_combine<<<divup((long long)NN * 32, TB), TB, 0, stream>>>(agg, hlin, deginv, b2, h2, (long long)NN * 32, 32);
  k_f32_to_f16<<<divup((long long)NN * 32, TB), TB, 0, stream>>>(h2, h2f16, (long long)NN * 32);

  // ---- GCN layer 3: x2 @ W3
  k_gemm_wmma<<<divup(6250 * 2, 8), TB, 0, stream>>>(h2f16, w3f, hlin, nullptr, NN, 32, 1, 2, 6250, 32, 0);
  k_fill_f32<<<divup((long long)NN * 32, TB), TB, 0, stream>>>(agg, (long long)NN * 32, 0.0f);
  k_scatter32<<<divup((long long)NE * 8, TB), TB, 0, stream>>>(src, dst, dis, hlin, agg, NE);
  k_combine<<<divup((long long)NN * 32, TB), TB, 0, stream>>>(agg, hlin, deginv, b3, h3, (long long)NN * 32, 32);

  // ---- GCN layer 4 (consumes x2, out_dim=1)
  k_lin32to1<<<divup(NN, TB), TB, 0, stream>>>(h2, W4, hlin1, NN);
  k_fill_f32<<<divup(NN, TB), TB, 0, stream>>>(agg1, NN, 0.0f);
  k_scatter1<<<divup(NE, TB), TB, 0, stream>>>(src, dst, dis, hlin1, agg1, NE);
  k_combine<<<divup(NN, TB), TB, 0, stream>>>(agg1, hlin1, deginv, b4, h4, NN, 1);

  // ---- global sort pooling (also emits f16 GEMM-A with K padded 97->128)
  k_sortpool<<<NG, 512, 0, stream>>>(h1, h2, h3, h4, xs16);

  // ---- conv1 (per-slot linear): [25000,97] @ [97,128], relu+bias
  k_gemm_wmma<<<divup(1563 * 8, 8), TB, 0, stream>>>(xs16, cw1f, c1, cb1, NG * KTOP, 128, 4, 8, 1563, 128, 1);

  // ---- maxpool(2,2) -> f16 [b][j][c]
  k_pool<<<divup((long long)NG * 50 * 128, TB), TB, 0, stream>>>(c1, p1f16, (long long)NG * 50 * 128);

  // ---- conv2: [11500,640] @ [640,64], relu+bias, permuted store -> [B, o*46+h]
  k_gemm_wmma<<<divup(719 * 4, 8), TB, 0, stream>>>(p1f16, cw2f, c2, cb2, NG * 46, 0, 20, 4, 719, 0, 2);

  // ---- fc1: [250,2944] @ [2944,128], relu+bias
  k_f32_to_f16<<<divup((long long)NG * DENSE, TB), TB, 0, stream>>>(c2, c2f16, (long long)NG * DENSE);
  k_gemm_wmma<<<divup(16 * 8, 8), TB, 0, stream>>>(c2f16, fw1f, fco, fb1, NG, DENSE, 92, 8, 16, GDIM, 1);

  // ---- fc3 + log_softmax
  k_fc3<<<1, TB, 0, stream>>>(fco, fw3, fb3, out);
}
